// ModulatedConv2D_58007828299865
// MI455X (gfx1250) — matrix operand order repaired and checked by reading, not compiled, not run on hardware
//
#include <hip/hip_runtime.h>

// ---------------------------------------------------------------------------
// ModulatedConv2D for MI455X (gfx1250), bf16 WMMA path with f32 accumulation.
// B=16, H=W=64, Cin=Cout=256, 3x3 SAME.  77.3 GFLOP vs ~170 MB -> compute
// bound; bf16 16x16x32 WMMA is 8x denser than the f32 16x16x4 path.
// Async global->LDS staging (ASYNCcnt) + LDS double buffering.
// ---------------------------------------------------------------------------

typedef __bf16 bf16;
typedef __attribute__((ext_vector_type(16))) __bf16 v16bf;
typedef __attribute__((ext_vector_type(8)))  __bf16 v8bf;
typedef __attribute__((ext_vector_type(4)))  __bf16 v4bf;
typedef __attribute__((ext_vector_type(2)))  __bf16 v2bf;
typedef __attribute__((ext_vector_type(8)))  float  v8f;
typedef __attribute__((ext_vector_type(4)))  int    i32x4;

// builtin signature: (int4 addrspace(1)* src, int4 addrspace(3)* dst, Imm, Imm)
typedef __attribute__((address_space(1))) i32x4 gas_i32x4;
typedef __attribute__((address_space(3))) i32x4 las_i32x4;

#define NB   16
#define HH   64
#define WW   64
#define CIN  256
#define COUT 256
#define TAPS 9
#define LDSC 66   // 64 cols + 1-col zero halo each side
#define SBUF (4 * LDSC * 32)   // bf16 elems per stage buffer

// eq = sqrt(2/(3*3*256)), eq^2 = 2/2304
#define EQ_F  0.0294627825f
#define EQ2_F 8.68055556e-4f

#if __has_builtin(__builtin_amdgcn_global_load_async_to_lds_b128)
#define ASYNC_LDS 1
#else
#define ASYNC_LDS 0
#endif

__device__ __forceinline__ void async_cp16(const bf16* g, bf16* l) {
#if ASYNC_LDS
  // 16-byte per-lane async copy, memory -> LDS, no VGPR round trip (ASYNCcnt)
  __builtin_amdgcn_global_load_async_to_lds_b128((gas_i32x4*)g, (las_i32x4*)l, 0, 0);
#else
  *(v8bf*)l = *(const v8bf*)g;    // synchronous fallback through VGPRs
#endif
}

__device__ __forceinline__ void wait_async() {
#if ASYNC_LDS
#if __has_builtin(__builtin_amdgcn_s_wait_asynccnt)
  __builtin_amdgcn_s_wait_asynccnt(0);
#else
  asm volatile("s_wait_asynccnt 0x0" ::: "memory");
#endif
#endif
}

// ---------------------------------------------------------------------------
// Kernel 0: image fp32 -> bf16 copy (same NHWC layout) so conv staging is a
// pure byte copy eligible for GLOBAL_LOAD_ASYNC_TO_LDS_B128.
// ---------------------------------------------------------------------------
__global__ void img2bf(const float* __restrict__ img, bf16* __restrict__ o) {
  const int i = (blockIdx.x * 256 + threadIdx.x) * 4;
  const float4 f = *(const float4*)(img + i);
  v4bf v; v[0] = (bf16)f.x; v[1] = (bf16)f.y; v[2] = (bf16)f.z; v[3] = (bf16)f.w;
  *(v4bf*)(o + i) = v;
}

// ---------------------------------------------------------------------------
// Kernel 1: demodulation sigma[b][cout] = rsqrt(eq^2 * sum (k*style)^2 + 1e-8)
// ---------------------------------------------------------------------------
__global__ void modsigma(const float* __restrict__ kernels,
                         const float* __restrict__ style,
                         float* __restrict__ sigma) {
  const int b = blockIdx.x;
  const int t = threadIdx.x;            // cout
  __shared__ float s2[CIN];
  float sv = style[b * CIN + t];
  s2[t] = sv * sv;
  __syncthreads();
  float acc = 0.f;
  for (int i = 0; i < TAPS * CIN; ++i) {   // i = tap*256 + cin
    float k = kernels[i * COUT + t];
    acc = fmaf(k * k, s2[i & (CIN - 1)], acc);
  }
  sigma[b * COUT + t] = rsqrtf(EQ2_F * acc + 1e-8f);
}

// ---------------------------------------------------------------------------
// Kernel 2: pack modulated+demodulated weights as bf16 in WMMA B-fragment
// layout in this code's contiguous-K order (32x16 KxN tile = 1024 B; lane = 16*(K/16)+N, elem e = K%16).
// Each thread writes one 4-byte K-pair; stores are fully contiguous.
// Fragment index fg = ((b*9 + tap)*8 + kb)*16 + nb.
// ---------------------------------------------------------------------------
__global__ void modpack(const float* __restrict__ kernels,
                        const float* __restrict__ style,
                        const float* __restrict__ sigma,
                        bf16* __restrict__ wq) {
  const int tid  = threadIdx.x;
  const int lane = tid >> 3;
  const int e    = (tid & 7) * 2;
  const int h    = lane >> 4;
  const int n16  = lane & 15;
  const int k32  = h * 16 + e;          // B layout: K = 16h + e
  for (int fg = blockIdx.x * 8; fg < blockIdx.x * 8 + 8; ++fg) {
    const int nb  = fg & 15;
    const int kb  = (fg >> 4) & 7;
    const int tap = (fg >> 7) % 9;
    const int bb  = fg / (16 * 8 * 9);
    const int cin0 = kb * 32 + k32;
    const int cout = nb * 16 + n16;
    const float sc = EQ_F * sigma[bb * COUT + cout];
    const float w0 = kernels[(tap * CIN + cin0    ) * COUT + cout] * style[bb * CIN + cin0    ] * sc;
    const float w1 = kernels[(tap * CIN + cin0 + 1) * COUT + cout] * style[bb * CIN + cin0 + 1] * sc;
    v2bf o; o[0] = (bf16)w0; o[1] = (bf16)w1;
    *(v2bf*)(wq + (size_t)fg * 512 + tid * 2) = o;
  }
}

// ---------------------------------------------------------------------------
// Kernel 3: implicit-GEMM conv. Block 256 = 8 wave32, tile 128M x 64N,
// wave -> 32x32 = 2x2 v_wmma_f32_16x16x32_bf16 accumulators.
// K loop: 8 cin-blocks x 9 taps; image staged per cin-block into one of two
// LDS buffers via async global->LDS copies overlapping the WMMA block.
// Halo cols / OOB rows are zeroed once and never rewritten.
// ---------------------------------------------------------------------------
__global__ __launch_bounds__(256) void modconv(const bf16* __restrict__ imgbf,
                                               const bf16* __restrict__ wq,
                                               float* __restrict__ out) {
  __shared__ bf16 simg[2][SBUF];                // 2 x 16,896 B

  const int tid   = threadIdx.x;
  const int lane  = tid & 31;
  const int wav   = tid >> 5;
  const int waveM = wav & 3;
  const int waveN = wav >> 2;
  const int mtile = blockIdx.x;                 // 0..31 (128 pixels)
  const int ntile = blockIdx.y;                 // 0..3  (64 couts)
  const int b     = blockIdx.z;
  const int y0    = mtile * 2;

  const int h      = lane >> 4;
  const int n16l   = lane & 15;
  const int prBase = waveM * 32 + n16l;         // A frag: M = lane%16
  const int nbBase = ntile * 4 + waveN * 2;     // first B-frag N index

  // ---- zero both buffers once (persistent zero halo) ----
  {
    v8bf z;
#pragma unroll
    for (int e = 0; e < 8; ++e) z[e] = (bf16)0.f;
    bf16* flat = &simg[0][0];
    for (int u = tid; u < 2 * SBUF / 8; u += 256) *(v8bf*)(flat + u * 8) = z;
  }
  __syncthreads();

  // ---- stage: async-copy interior cells of rows [y0-1,y0+2] for cin-block kb
  auto stage = [&](int kb, int buf) {
    for (int u = tid; u < 4 * 64 * 4; u += 256) {   // (r, gx, 16B-chunk)
      const int q2 = u & 3;
      const int gx = (u >> 2) & 63;
      const int r  = u >> 8;
      const int gy = y0 - 1 + r;
      if ((unsigned)gy < 64u) {
        const bf16* g = imgbf + (size_t)((b * HH + gy) * WW + gx) * CIN + kb * 32 + q2 * 8;
        bf16* l = &simg[buf][(r * LDSC + (gx + 1)) * 32 + q2 * 8];
        async_cp16(g, l);
      }
    }
  };

  v8f acc[2][2] = {};

  stage(0, 0);
  wait_async();
  __syncthreads();

  for (int kb = 0; kb < 8; ++kb) {
    const int cur = kb & 1;
    if (kb < 7) {
      stage(kb + 1, cur ^ 1);                      // overlap copy with compute
      const char* pf = (const char*)wq +
          (size_t)(((b * 9) * 8 + (kb + 1)) * 16 + nbBase) * 1024 + lane * 32;
      __builtin_prefetch(pf, 0, 1);                // global_prefetch_b8
    }

    const bf16* sb = &simg[cur][0];
    for (int tap = 0; tap < 9; ++tap) {
      const int dy = tap / 3, dx = tap % 3;
      const size_t fb = (size_t)(((b * 9 + tap) * 8 + kb) * 16 + nbBase);
      union { v16bf v; v8bf hv[2]; } Bf[2], Af[2];
      Bf[0].v = *(const v16bf*)(wq + fb * 512 + lane * 16);
      Bf[1].v = *(const v16bf*)(wq + (fb + 1) * 512 + lane * 16);
#pragma unroll
      for (int f = 0; f < 2; ++f) {
        const int pr  = prBase + 16 * f;
        const int row = (pr >> 6) + dy;
        const int col = (pr & 63) + dx;
        const bf16* ap = sb + (row * LDSC + col) * 32 + 16 * h;
        Af[f].hv[0] = *(const v8bf*)ap;            // K = 16h..16h+7
        Af[f].hv[1] = *(const v8bf*)(ap + 8);      // K = 16h+8..16h+15
      }
#pragma unroll
      for (int i = 0; i < 2; ++i)
#pragma unroll
        for (int j = 0; j < 2; ++j)
          acc[i][j] = __builtin_amdgcn_wmma_f32_16x16x32_bf16(
              false, Af[i].v, false, Bf[j].v, (short)0, acc[i][j], false, false);
    }

    if (kb < 7) {
      wait_async();                                // this wave's copies done
      __syncthreads();                             // everyone's copies visible
    }
  }

  // ---- writeback: C/D layout lane L -> N = L%16, M = v + 8*(L/16) ----
  const int pixBase  = mtile * 128 + waveM * 32;
  const int coutBase = ntile * 64 + waveN * 32 + n16l;
#pragma unroll
  for (int i = 0; i < 2; ++i)
#pragma unroll
    for (int j = 0; j < 2; ++j)
#pragma unroll
      for (int v = 0; v < 8; ++v) {
        const int pix = pixBase + 16 * i + v + 8 * h;
        const int co  = coutBase + 16 * j;
        out[((size_t)(b * 4096 + pix)) * COUT + co] = acc[i][j][v];
      }
}

// ---------------------------------------------------------------------------
extern "C" void kernel_launch(void* const* d_in, const int* in_sizes, int n_in,
                              void* d_out, int out_size, void* d_ws, size_t ws_size,
                              hipStream_t stream) {
  const float* image   = (const float*)d_in[0];   // (16,64,64,256)
  const float* style   = (const float*)d_in[1];   // (16,256)
  const float* kernels = (const float*)d_in[2];   // (3,3,256,256)
  float* out = (float*)d_out;

  // ws layout: [0,16KB) sigma f32; [16KB,+18.87MB) packed bf16 weights;
  //            then 33.55MB bf16 image copy.  total ~52.4 MB.
  float* sigma = (float*)d_ws;
  bf16*  wq    = (bf16*)((char*)d_ws + 16384);
  bf16*  imgbf = (bf16*)((char*)d_ws + 16384 + (size_t)(NB * TAPS * 8 * 16) * 1024);

  img2bf<<<(NB * HH * WW * CIN) / (256 * 4), 256, 0, stream>>>(image, imgbf);
  modsigma<<<NB, 256, 0, stream>>>(kernels, style, sigma);
  modpack<<<(NB * TAPS * 8 * 16) / 8, 256, 0, stream>>>(kernels, style, sigma, wq);
  dim3 grid(32, 4, NB);
  modconv<<<grid, 256, 0, stream>>>(imgbf, wq, out);
}
